// GroupedQueryAttention_2817498546641
// MI455X (gfx1250) — matrix operand order repaired and checked by reading, not compiled, not run on hardware
//
#include <hip/hip_runtime.h>
#include <stdint.h>

typedef __bf16 bf16;
typedef bf16  v16bf __attribute__((ext_vector_type(16)));
typedef bf16  v8bf  __attribute__((ext_vector_type(8)));
typedef float v8f   __attribute__((ext_vector_type(8)));

#define B_  2
#define S_  2048
#define D_  2048
#define H_  16
#define KV_ 4
#define HD_ 128

union Afrag { v16bf v; v8bf h[2]; };

static __device__ __forceinline__ v8f wmma_bf16(v16bf a, v16bf b, v8f c) {
  // D = A(16x32 bf16) x B(32x16 bf16) + C(16x16 f32)
  return __builtin_amdgcn_wmma_f32_16x16x32_bf16(false, a, false, b, (short)0, c, false, false);
}

static __device__ __forceinline__ v8f zero8() {
  v8f z = {0.f,0.f,0.f,0.f,0.f,0.f,0.f,0.f};
  return z;
}

// Async global->LDS copy of a 32x128 bf16 tile (row stride HD_ in global), 8KB.
// 16 instructions; each moves 32 lanes x 16B = 512B (2 rows).
static __device__ __forceinline__ void async_copy_k(const bf16* gp, bf16* lp, int lane) {
  unsigned lbase = (unsigned)(uintptr_t)lp;
  int sub = lane & 15;
  #pragma unroll
  for (int i = 0; i < 16; ++i) {
    int row = i * 2 + (lane >> 4);
    unsigned laddr = lbase + (unsigned)(row * 256 + sub * 16);
    unsigned long long ga = (unsigned long long)(uintptr_t)(gp + (size_t)row * HD_ + sub * 8);
    asm volatile("global_load_async_to_lds_b128 %0, %1, off"
                 :: "v"(laddr), "v"(ga) : "memory");
  }
}

// Async global->LDS copy of a 128x32 bf16 tile (row stride S_ in global), 8KB.
// 16 instructions; each moves 8 rows of 64B.
static __device__ __forceinline__ void async_copy_v(const bf16* gp, bf16* lp, int lane) {
  unsigned lbase = (unsigned)(uintptr_t)lp;
  int sub = lane & 3;
  #pragma unroll
  for (int i = 0; i < 16; ++i) {
    int row = i * 8 + (lane >> 2);
    unsigned laddr = lbase + (unsigned)(row * 64 + sub * 16);
    unsigned long long ga = (unsigned long long)(uintptr_t)(gp + (size_t)row * S_ + sub * 8);
    asm volatile("global_load_async_to_lds_b128 %0, %1, off"
                 :: "v"(laddr), "v"(ga) : "memory");
  }
}

// ---------------- elementwise converts ----------------

__global__ void cvt_bf16_kernel(const float* __restrict__ in, bf16* __restrict__ out, int n) {
  int i = blockIdx.x * blockDim.x + threadIdx.x;
  if (i < n) out[i] = (bf16)in[i];
}

// W [K,N] f32 row-major -> Wt [N,K] bf16 row-major (coalesced writes)
__global__ void transpose_cvt_kernel(const float* __restrict__ in, bf16* __restrict__ out,
                                     int K, int N) {
  int i = blockIdx.x * blockDim.x + threadIdx.x;
  if (i >= N * K) return;
  int n = i / K;
  int k = i - n * K;
  out[i] = (bf16)in[(size_t)k * N + n];
}

// ---------------- WMMA GEMM: C[M,N] = A[M,K] * Bt[N,K]^T ----------------
// 1 wave computes a 16(M) x 64(N) tile; K-chunks of 32.
template<int OUTF32>
__global__ __launch_bounds__(32) void gemm_wmma(const bf16* __restrict__ A,
                                                const bf16* __restrict__ Bt,
                                                void* __restrict__ Cout,
                                                int M, int N, int K) {
  int n0 = blockIdx.x * 64;
  int m0 = blockIdx.y * 16;
  int lane = threadIdx.x;
  int hf = lane >> 4;     // half of wave
  int lo = lane & 15;     // row (A) / column (B,C)
  v8f acc[4];
  #pragma unroll
  for (int t = 0; t < 4; ++t) acc[t] = zero8();
  const bf16* arow = A + (size_t)(m0 + lo) * K;
  for (int kc = 0; kc < K; kc += 32) {
    Afrag a;
    a.h[0] = *(const v8bf*)(arow + kc + 16 * hf);        // K in [16h, 16h+8)
    a.h[1] = *(const v8bf*)(arow + kc + 16 * hf + 8);    // K in [16h+8, 16h+16)
    __builtin_prefetch(arow + kc + 256, 0, 0);           // global_prefetch_b8
    #pragma unroll
    for (int t = 0; t < 4; ++t) {
      const bf16* brow = Bt + (size_t)(n0 + t * 16 + lo) * K;
      v16bf bfr = *(const v16bf*)(brow + kc + 16 * hf);  // K in [16h, 16h+16)
      acc[t] = wmma_bf16(a.v, bfr, acc[t]);
    }
  }
  #pragma unroll
  for (int t = 0; t < 4; ++t) {
    #pragma unroll
    for (int r = 0; r < 8; ++r) {
      int row = m0 + r + 8 * hf;
      int col = n0 + t * 16 + lo;
      if (OUTF32) ((float*)Cout)[(size_t)row * N + col] = acc[t][r];
      else        ((bf16*)Cout)[(size_t)row * N + col] = (bf16)acc[t][r];
    }
  }
}

// ---------------- RoPE + relayout ----------------
// q_lin [B,S,H*HD] bf16 -> q_r [B,H,S,HD] bf16 with rope
__global__ void rope_q_kernel(const bf16* __restrict__ qlin, const float* __restrict__ cosp,
                              const float* __restrict__ sinp, bf16* __restrict__ qout) {
  int i = blockIdx.x * blockDim.x + threadIdx.x;  // B*S*H*64 = 2^22
  if (i >= B_ * S_ * H_ * (HD_ / 2)) return;
  int p = i & 63;
  int h = (i >> 6) & (H_ - 1);
  int s = (i >> 10) & (S_ - 1);
  int b = i >> 21;
  float c  = cosp[s * (HD_ / 2) + p];
  float sn = sinp[s * (HD_ / 2) + p];
  size_t inoff = ((size_t)(b * S_ + s)) * (H_ * HD_) + h * HD_ + 2 * p;
  float e = (float)qlin[inoff];
  float o = (float)qlin[inoff + 1];
  size_t outoff = (((size_t)(b * H_ + h)) * S_ + s) * HD_ + 2 * p;
  qout[outoff]     = (bf16)(e * c - o * sn);
  qout[outoff + 1] = (bf16)(e * sn + o * c);
}

// k_lin [B,S,KV*HD] -> k_r [B,KV,S,HD] with rope
__global__ void rope_k_kernel(const bf16* __restrict__ klin, const float* __restrict__ cosp,
                              const float* __restrict__ sinp, bf16* __restrict__ kout) {
  int i = blockIdx.x * blockDim.x + threadIdx.x;  // B*S*KV*64 = 2^20
  if (i >= B_ * S_ * KV_ * (HD_ / 2)) return;
  int p = i & 63;
  int g = (i >> 6) & (KV_ - 1);
  int s = (i >> 8) & (S_ - 1);
  int b = i >> 19;
  float c  = cosp[s * (HD_ / 2) + p];
  float sn = sinp[s * (HD_ / 2) + p];
  size_t inoff = ((size_t)(b * S_ + s)) * (KV_ * HD_) + g * HD_ + 2 * p;
  float e = (float)klin[inoff];
  float o = (float)klin[inoff + 1];
  size_t outoff = (((size_t)(b * KV_ + g)) * S_ + s) * HD_ + 2 * p;
  kout[outoff]     = (bf16)(e * c - o * sn);
  kout[outoff + 1] = (bf16)(e * sn + o * c);
}

// v_lin [B,S,KV*HD] -> v_t [B,KV,HD,S]  (d-major so P*V B-frags are contiguous)
__global__ void vtrans_kernel(const bf16* __restrict__ vlin, bf16* __restrict__ vt) {
  int i = blockIdx.x * blockDim.x + threadIdx.x;  // B*KV*HD*S = 2^21
  if (i >= B_ * KV_ * HD_ * S_) return;
  int s = i & (S_ - 1);
  int d = (i >> 11) & (HD_ - 1);
  int g = (i >> 18) & (KV_ - 1);
  int b = i >> 20;
  vt[i] = vlin[((size_t)(b * S_ + s)) * (KV_ * HD_) + g * HD_ + d];
}

// ---------------- Flash attention (causal, GQA) ----------------
// 1 wave per 16-query tile; 32 keys/iter; K/V tiles double-buffered in LDS via
// global_load_async_to_lds_b128 (ASYNCcnt), online softmax, WMMA for S=QK^T and O=PV.
__global__ __launch_bounds__(32) void flash_attn_kernel(const bf16* __restrict__ Q,
                                                        const bf16* __restrict__ Kr,
                                                        const bf16* __restrict__ Vt,
                                                        bf16* __restrict__ O) {
  __shared__ __align__(16) bf16 kTile[2][32 * HD_];   // 2 x 8KB, row-major [32][128]
  __shared__ __align__(16) bf16 vTile[2][HD_ * 32];   // 2 x 8KB, d-major   [128][32]
  __shared__ __align__(16) bf16 pLDS[16 * 40];        // 16x32 P tile, row stride 40
  int qt = blockIdx.x, h = blockIdx.y, b = blockIdx.z;
  int g = h >> 2;  // kv group (N_REP = 4)
  int lane = threadIdx.x, hf = lane >> 4, nr = lane & 15;
  const bf16* qp = Q  + ((size_t)(b * H_ + h)) * S_ * HD_;
  const bf16* kp = Kr + ((size_t)(b * KV_ + g)) * S_ * HD_;
  const bf16* vp = Vt + ((size_t)(b * KV_ + g)) * HD_ * S_;
  int qbase = qt * 16;

  // Q A-fragments for the 4 HD chunks of 32 (kept in registers for whole kernel)
  Afrag qa[4];
  #pragma unroll
  for (int c = 0; c < 4; ++c) {
    const bf16* qrow = qp + (size_t)(qbase + nr) * HD_ + 32 * c;
    qa[c].h[0] = *(const v8bf*)(qrow + 16 * hf);
    qa[c].h[1] = *(const v8bf*)(qrow + 16 * hf + 8);
  }

  v8f acc[8];
  #pragma unroll
  for (int t = 0; t < 8; ++t) acc[t] = zero8();
  float mrow[8], lrow[8];
  #pragma unroll
  for (int r = 0; r < 8; ++r) { mrow[r] = -__builtin_inff(); lrow[r] = 0.f; }
  const float scale = 0.08838834764831845f;  // 1/sqrt(128)

  int niter = (qbase + 16 + 31) >> 5;

  // prefetch tile 0 (32 async b128 instructions = one K tile + one V tile)
  async_copy_k(kp, &kTile[0][0], lane);
  async_copy_v(vp, &vTile[0][0], lane);

  for (int it = 0; it < niter; ++it) {
    int kb = it * 32;
    int buf = it & 1;
    if (it + 1 < niter) {
      // issue next tile's 32 async copies, then release current tile (in-order
      // completion: asynccnt <= 32 means the first 32 have landed in LDS)
      const bf16* kn = kp + (size_t)(kb + 32) * HD_;
      const bf16* vn = vp + (kb + 32);
      async_copy_k(kn, &kTile[buf ^ 1][0], lane);
      async_copy_v(vn, &vTile[buf ^ 1][0], lane);
      asm volatile("s_wait_asynccnt 32" ::: "memory");
    } else {
      asm volatile("s_wait_asynccnt 0" ::: "memory");
    }
    const bf16* kt = &kTile[buf][0];
    const bf16* vt = &vTile[buf][0];

    // scores: two 16x16 tiles over 32 keys, K/V fragments from LDS
    v8f s0 = zero8(), s1 = zero8();
    #pragma unroll
    for (int c = 0; c < 4; ++c) {
      v16bf b0 = *(const v16bf*)(kt + nr * HD_ + 32 * c + 16 * hf);
      v16bf b1 = *(const v16bf*)(kt + (16 + nr) * HD_ + 32 * c + 16 * hf);
      s0 = wmma_bf16(qa[c].v, b0, s0);
      s1 = wmma_bf16(qa[c].v, b1, s1);
    }
    // causal mask + online softmax (rows live across 16-lane halves)
    float p0[8], p1[8];
    #pragma unroll
    for (int r = 0; r < 8; ++r) {
      int qg = qbase + r + 8 * hf;
      float v0 = (kb + nr      <= qg) ? s0[r] * scale : -__builtin_inff();
      float v1 = (kb + 16 + nr <= qg) ? s1[r] * scale : -__builtin_inff();
      float mx = fmaxf(v0, v1);
      #pragma unroll
      for (int off = 8; off; off >>= 1) mx = fmaxf(mx, __shfl_xor(mx, off, 32));
      float mnew = fmaxf(mrow[r], mx);
      float e0 = __expf(v0 - mnew);
      float e1 = __expf(v1 - mnew);
      float sum = e0 + e1;
      #pragma unroll
      for (int off = 8; off; off >>= 1) sum += __shfl_xor(sum, off, 32);
      float alpha = __expf(mrow[r] - mnew);
      lrow[r] = lrow[r] * alpha + sum;
      mrow[r] = mnew;
      p0[r] = e0; p1[r] = e1;
      #pragma unroll
      for (int t = 0; t < 8; ++t) acc[t][r] *= alpha;
    }
    // transpose P (C-layout -> A-layout) through LDS
    #pragma unroll
    for (int r = 0; r < 8; ++r) {
      pLDS[(r + 8 * hf) * 40 + nr]      = (bf16)p0[r];
      pLDS[(r + 8 * hf) * 40 + 16 + nr] = (bf16)p1[r];
    }
    __syncthreads();
    Afrag pa;
    pa.h[0] = *(const v8bf*)&pLDS[nr * 40 + 16 * hf];
    pa.h[1] = *(const v8bf*)&pLDS[nr * 40 + 16 * hf + 8];
    // out += P(16x32) * V(32x128): 8 WMMAs, V B-frags contiguous in LDS
    #pragma unroll
    for (int t = 0; t < 8; ++t) {
      v16bf vb = *(const v16bf*)(vt + (t * 16 + nr) * 32 + 16 * hf);
      acc[t] = wmma_bf16(pa.v, vb, acc[t]);
    }
    __syncthreads();
  }
  // normalize and store to attn buffer [B,S,H*HD]
  #pragma unroll
  for (int t = 0; t < 8; ++t) {
    #pragma unroll
    for (int r = 0; r < 8; ++r) {
      int row = qbase + r + 8 * hf;
      float o = acc[t][r] / lrow[r];
      O[((size_t)b * S_ + row) * (H_ * HD_) + h * HD_ + t * 16 + nr] = (bf16)o;
    }
  }
}

// ---------------- launch ----------------

extern "C" void kernel_launch(void* const* d_in, const int* in_sizes, int n_in,
                              void* d_out, int out_size, void* d_ws, size_t ws_size,
                              hipStream_t stream) {
  const float* x  = (const float*)d_in[0];
  const float* rc = (const float*)d_in[1];
  const float* rs = (const float*)d_in[2];
  const float* Wq = (const float*)d_in[3];
  const float* Wk = (const float*)d_in[4];
  const float* Wv = (const float*)d_in[5];
  const float* Wo = (const float*)d_in[6];
  float* out = (float*)d_out;

  char* w = (char*)d_ws;
  const size_t MB = 1024 * 1024;
  // workspace layout (100 MB total)
  bf16* xb    = (bf16*)(w);            // x bf16            [4096,2048]   16 MB
  bf16* wq_t  = (bf16*)(w + 16 * MB);  // Wq^T bf16         [2048,2048]    8 MB
  bf16* wk_t  = (bf16*)(w + 24 * MB);  // Wk^T bf16         [512,2048]     2 MB
  bf16* wv_t  = (bf16*)(w + 26 * MB);  // Wv^T bf16         [512,2048]     2 MB
  bf16* wo_t  = (bf16*)(w + 28 * MB);  // Wo^T bf16         [2048,2048]    8 MB
  bf16* q_lin = (bf16*)(w + 36 * MB);  // x@Wq bf16         [4096,2048]   16 MB
  bf16* k_lin = (bf16*)(w + 52 * MB);  // x@Wk bf16         [4096,512]     4 MB
  bf16* v_lin = (bf16*)(w + 56 * MB);  // x@Wv bf16         [4096,512]     4 MB
  bf16* q_r   = (bf16*)(w + 60 * MB);  // rope(q)           [B,H,S,HD]    16 MB
  bf16* k_r   = (bf16*)(w + 76 * MB);  // rope(k)           [B,KV,S,HD]    4 MB
  bf16* v_t   = (bf16*)(w + 80 * MB);  // v d-major         [B,KV,HD,S]    4 MB
  bf16* attn  = (bf16*)(w + 84 * MB);  // attention out     [B,S,H*HD]    16 MB

  const int M = B_ * S_;  // 4096

  // stage 0: converts / transposes
  {
    int n = M * D_;
    cvt_bf16_kernel<<<(n + 255) / 256, 256, 0, stream>>>(x, xb, n);
  }
  transpose_cvt_kernel<<<(D_ * D_ + 255) / 256, 256, 0, stream>>>(Wq, wq_t, D_, D_);
  transpose_cvt_kernel<<<(D_ * 512 + 255) / 256, 256, 0, stream>>>(Wk, wk_t, D_, 512);
  transpose_cvt_kernel<<<(D_ * 512 + 255) / 256, 256, 0, stream>>>(Wv, wv_t, D_, 512);
  transpose_cvt_kernel<<<(D_ * D_ + 255) / 256, 256, 0, stream>>>(Wo, wo_t, D_, D_);

  // stage 1: QKV projections (WMMA)
  gemm_wmma<0><<<dim3(D_ / 64, M / 16), 32, 0, stream>>>(xb, wq_t, (void*)q_lin, M, D_, D_);
  gemm_wmma<0><<<dim3(512 / 64, M / 16), 32, 0, stream>>>(xb, wk_t, (void*)k_lin, M, 512, D_);
  gemm_wmma<0><<<dim3(512 / 64, M / 16), 32, 0, stream>>>(xb, wv_t, (void*)v_lin, M, 512, D_);

  // stage 2: rope + relayout
  rope_q_kernel<<<(B_ * S_ * H_ * 64 + 255) / 256, 256, 0, stream>>>(q_lin, rc, rs, q_r);
  rope_k_kernel<<<(B_ * S_ * KV_ * 64 + 255) / 256, 256, 0, stream>>>(k_lin, rc, rs, k_r);
  vtrans_kernel<<<(B_ * KV_ * HD_ * S_ + 255) / 256, 256, 0, stream>>>(v_lin, v_t);

  // stage 3: causal flash attention (WMMA + async LDS double-buffering)
  flash_attn_kernel<<<dim3(S_ / 16, H_, B_), 32, 0, stream>>>(q_r, k_r, v_t, attn);

  // stage 4: output projection, fp32 result (WMMA)
  gemm_wmma<1><<<dim3(D_ / 64, M / 16), 32, 0, stream>>>(attn, wo_t, (void*)out, M, D_, D_);
}
